// WindowAttention_19327352832689
// MI455X (gfx1250) — compile-verified, matrix operand order checked by reading, not run on hardware
//
#include <hip/hip_runtime.h>

typedef __attribute__((ext_vector_type(16))) _Float16 v16h;
typedef __attribute__((ext_vector_type(8)))  float    v8f;

#define WA_N     49
#define WA_NP    64
#define WA_DIM   128
#define WA_HEADS 4
#define WA_HD    32
#define WA_NW    1024
#define WA_SCALE 0.17677669529663687f   // 32^-0.5

#define WQKV_ELEMS (3 * WA_DIM * WA_DIM)   // 49152
#define WPROJ_ELEMS (WA_DIM * WA_DIM)      // 16384
#define WS_F16_BYTES ((WQKV_ELEMS + WPROJ_ELEMS) * 2)  // 131072
#define WA_LDS_BYTES 98304

// ---- WMMA f16 -> f32 (D = A*B + C), 16x16x32, wave32 ----
__device__ __forceinline__ v8f wmma_f16(v16h a, v16h b, v8f c) {
  return __builtin_amdgcn_wmma_f32_16x16x32_f16(
      /*neg_a=*/false, a, /*neg_b=*/false, b,
      /*c_mod=*/(short)0, c, /*reuse_a=*/false, /*reuse_b=*/false);
}

// A/B operand from f16 row-major [rows][stride] (LDS or global):
// lane supplies row (A: M-row, B: N-column); K elements are two contiguous
// 8xf16 chunks at kb and kb+16 (kb already includes the lane-half offset).
__device__ __forceinline__ v16h f16_oper(const _Float16* __restrict__ base,
                                         int row, int stride, int kb) {
  union { v16h v; float4 q[2]; } u;
  const _Float16* p = base + row * stride + kb;
  u.q[0] = *(const float4*)(p);
  u.q[1] = *(const float4*)(p + 16);
  return u.v;
}

// B operand from a global f32 row-major weight matrix with on-the-fly
// conversion (fallback path when workspace is too small to stage f16 weights).
__device__ __forceinline__ v16h glb_oper_f32(const float* __restrict__ W,
                                             int row, int stride, int kb) {
  const float* p = W + row * stride + kb;
  float4 a0 = *(const float4*)(p);
  float4 a1 = *(const float4*)(p + 4);
  float4 a2 = *(const float4*)(p + 16);
  float4 a3 = *(const float4*)(p + 20);
  v16h r;
  r[0]  = (_Float16)a0.x; r[1]  = (_Float16)a0.y; r[2]  = (_Float16)a0.z; r[3]  = (_Float16)a0.w;
  r[4]  = (_Float16)a1.x; r[5]  = (_Float16)a1.y; r[6]  = (_Float16)a1.z; r[7]  = (_Float16)a1.w;
  r[8]  = (_Float16)a2.x; r[9]  = (_Float16)a2.y; r[10] = (_Float16)a2.z; r[11] = (_Float16)a2.w;
  r[12] = (_Float16)a3.x; r[13] = (_Float16)a3.y; r[14] = (_Float16)a3.z; r[15] = (_Float16)a3.w;
  return r;
}

// ---- one-shot weight conversion: qkv_w then proj_w, f32 -> f16 in ws ----
__global__ __launch_bounds__(256)
void convert_weights_kernel(const float* __restrict__ qkv_w,
                            const float* __restrict__ proj_w,
                            _Float16* __restrict__ dst) {
  int i4 = (blockIdx.x * 256 + threadIdx.x) * 4;   // grid covers 65536 elems
  float4 v = (i4 < WQKV_ELEMS) ? *(const float4*)(qkv_w + i4)
                               : *(const float4*)(proj_w + (i4 - WQKV_ELEMS));
  union { _Float16 h[4]; float2 f; } u;
  u.h[0] = (_Float16)v.x; u.h[1] = (_Float16)v.y;
  u.h[2] = (_Float16)v.z; u.h[3] = (_Float16)v.w;
  *(float2*)(dst + i4) = u.f;
}

template <bool WF16>
__global__ __launch_bounds__(256, 1)
void win_attn_fused_kernel(const float* __restrict__ x,
                           const float* __restrict__ mask,
                           const float* __restrict__ qkv_w32,
                           const _Float16* __restrict__ qkv_w16,
                           const float* __restrict__ qkv_b,
                           const float* __restrict__ proj_w32,
                           const _Float16* __restrict__ proj_w16,
                           const float* __restrict__ proj_b,
                           const float* __restrict__ rel_tab,
                           const int*   __restrict__ rel_idx,
                           float* __restrict__ out) {
  extern __shared__ unsigned char smem[];
  // LDS plan (96 KB total, lifetime-based overlays):
  _Float16* XH   = (_Float16*)(smem);           // [64][128] f16 x     (dead after QKV)
  _Float16* QH   = (_Float16*)(smem + 16384);   // [4][64][32] q*scale (dead after S)
  _Float16* PH   = (_Float16*)(smem);           // [4][64][64] softmax(P), overlays XH+QH
  _Float16* KH   = (_Float16*)(smem + 32768);   // [4][64][32] k       (dead after S)
  _Float16* OH   = (_Float16*)(smem + 32768);   // [64][128] attn out, overlays KH
  _Float16* VTH  = (_Float16*)(smem + 49152);   // [4][32][64] v^T (padded tokens = v bias)
  _Float16* BIAS = (_Float16*)(smem + 65536);   // [64][64][4] bias+mask, -inf for n>=49

  const int tid   = threadIdx.x;
  const int lane  = tid & 31;
  const int wvs   = __builtin_amdgcn_readfirstlane(tid >> 5);  // scalar wave id
  const int colT  = lane & 15;          // tile row/col selector
  const int lhalf = (lane >> 4) << 3;   // 0|8: C-row offset AND operand K offset

  const int bwin = blockIdx.x;
  const float* xw = x    + (size_t)bwin * (WA_N * WA_DIM);
  const float* mw = mask + (size_t)(bwin % WA_NW) * (WA_N * WA_N);

  // ---------- stage 1a: x -> LDS f16, rows 49..63 zeroed ----------
  for (int i = tid; i < (WA_NP * WA_DIM) / 4; i += 256) {
    int row = i >> 5;
    int c4  = (i & 31) << 2;
    float4 v = make_float4(0.f, 0.f, 0.f, 0.f);
    if (row < WA_N) v = *(const float4*)(xw + row * WA_DIM + c4);
    union { _Float16 h[4]; float2 f; } u;
    u.h[0] = (_Float16)v.x; u.h[1] = (_Float16)v.y;
    u.h[2] = (_Float16)v.z; u.h[3] = (_Float16)v.w;
    *(float2*)(XH + row * WA_DIM + c4) = u.f;
  }

  // ---------- stage 1b: stage bias+mask grid [64][64][4] into LDS ----------
  // cols >= 49 get f16 -inf (softmax pad); rows >= 49 use clamped values so
  // no row is all -inf (keeps padded-row softmax finite).
  for (int p = tid; p < WA_NP * WA_NP; p += 256) {
    int m = p >> 6, n = p & 63;
    int mm = m < WA_N ? m : WA_N - 1;
    union { _Float16 h[4]; float2 f; } u;
    if (n < WA_N) {
      int idx = rel_idx[mm * WA_N + n];
      float4 t4 = *(const float4*)(rel_tab + idx * WA_HEADS);
      float mv = mw[mm * WA_N + n];
      u.h[0] = (_Float16)(t4.x + mv); u.h[1] = (_Float16)(t4.y + mv);
      u.h[2] = (_Float16)(t4.z + mv); u.h[3] = (_Float16)(t4.w + mv);
    } else {
      _Float16 ninf = (_Float16)(-3.0e38f);   // -> -inf in f16
      u.h[0] = ninf; u.h[1] = ninf; u.h[2] = ninf; u.h[3] = ninf;
    }
    *(float2*)(BIAS + p * 4) = u.f;
  }
  __syncthreads();

  // ---------- stage 2: fused QKV projection [64 x 384 x 128] ----------
#pragma unroll 1
  for (int j = 0; j < 12; ++j) {
    const int t  = wvs + 8 * j;        // scalar
    const int mt = t & 3, nt = t >> 2; // scalar
    v8f acc = {};
#pragma unroll
    for (int ks = 0; ks < 4; ++ks) {
      v16h a = f16_oper(XH, mt * 16 + colT, WA_DIM, ks * 32 + lhalf);
      v16h b;
      if constexpr (WF16) b = f16_oper(qkv_w16, nt * 16 + colT, WA_DIM, ks * 32 + lhalf);
      else                b = glb_oper_f32(qkv_w32, nt * 16 + colT, WA_DIM, ks * 32 + lhalf);
      acc = wmma_f16(a, b, acc);
    }
    const int which = nt >> 3;          // scalar: 0=q 1=k 2=v
    const int h     = (nt >> 1) & 3;    // scalar head
    const int d     = ((nt & 1) << 4) + colT;
    const int m0    = mt * 16 + lhalf;
    float bb = qkv_b[nt * 16 + colT];
    if (which == 0) {
      _Float16* q = QH + h * 2048 + m0 * 32 + d;
#pragma unroll
      for (int i = 0; i < 8; ++i) q[i * 32] = (_Float16)((acc[i] + bb) * WA_SCALE);
    } else if (which == 1) {
      _Float16* k = KH + h * 2048 + m0 * 32 + d;
#pragma unroll
      for (int i = 0; i < 8; ++i) k[i * 32] = (_Float16)(acc[i] + bb);
    } else {
      // transposed V: 8 consecutive tokens -> one 16B LDS store
      union { _Float16 h8[8]; float4 f; } u;
#pragma unroll
      for (int i = 0; i < 8; ++i) u.h8[i] = (_Float16)(acc[i] + bb);
      *(float4*)(VTH + h * 2048 + d * 64 + m0) = u.f;
    }
  }
  __syncthreads();

  // ---------- stage 3: S = q@k^T + bias + mask (per head), softmax ----------
  const int h   = wvs >> 1;          // scalar: 2 waves per head
  const int mtb = (wvs & 1) * 2;     // scalar: this wave owns M-tiles {mtb, mtb+1}
  v8f S[2][4];
#pragma unroll
  for (int mi = 0; mi < 2; ++mi) {
    const int m0 = (mtb + mi) * 16 + lhalf;
#pragma unroll
    for (int nt = 0; nt < 4; ++nt) {
      v16h a = f16_oper(QH + h * 2048, (mtb + mi) * 16 + colT, WA_HD, lhalf); // K=32
      v16h b = f16_oper(KH + h * 2048, nt * 16 + colT, WA_HD, lhalf);
      v8f acc = {};
      acc = wmma_f16(a, b, acc);
      const _Float16* bp = BIAS + (m0 * 64 + nt * 16 + colT) * 4 + h;
#pragma unroll
      for (int i = 0; i < 8; ++i) acc[i] += (float)bp[i * 256];  // branch-free
      S[mi][nt] = acc;
    }
  }
  __syncthreads();   // all QH/KH reads complete before PH overlays them

  // register softmax; rows live per 16-lane half, reduce with xor-butterflies
#pragma unroll
  for (int mi = 0; mi < 2; ++mi) {
    const int mt = mtb + mi;
    float rmax[8], rsum[8];
#pragma unroll
    for (int i = 0; i < 8; ++i) {
      float mx = S[mi][0][i];
#pragma unroll
      for (int nt = 1; nt < 4; ++nt) mx = fmaxf(mx, S[mi][nt][i]);
#pragma unroll
      for (int off = 8; off >= 1; off >>= 1)
        mx = fmaxf(mx, __shfl_xor(mx, off, 32));
      rmax[i] = mx;
      rsum[i] = 0.f;
    }
#pragma unroll
    for (int nt = 0; nt < 4; ++nt)
#pragma unroll
      for (int i = 0; i < 8; ++i) {
        float e = __expf(S[mi][nt][i] - rmax[i]);
        S[mi][nt][i] = e;
        rsum[i] += e;
      }
#pragma unroll
    for (int i = 0; i < 8; ++i) {
      float s = rsum[i];
#pragma unroll
      for (int off = 8; off >= 1; off >>= 1) s += __shfl_xor(s, off, 32);
      rsum[i] = 1.0f / s;
    }
#pragma unroll
    for (int nt = 0; nt < 4; ++nt) {
      _Float16* pp = PH + h * 4096 + (mt * 16 + lhalf) * 64 + nt * 16 + colT;
#pragma unroll
      for (int i = 0; i < 8; ++i)
        pp[i * 64] = (_Float16)(S[mi][nt][i] * rsum[i]);
    }
  }
  __syncthreads();

  // ---------- stage 4: O = P @ V (K = 64 tokens, B = V^T rows) ----------
#pragma unroll
  for (int mi = 0; mi < 2; ++mi) {
    const int mt = mtb + mi;
#pragma unroll
    for (int nt = 0; nt < 2; ++nt) {
      v8f o = {};
#pragma unroll
      for (int ks = 0; ks < 2; ++ks) {
        v16h a = f16_oper(PH + h * 4096, mt * 16 + colT, 64, ks * 32 + lhalf);
        v16h b = f16_oper(VTH + h * 2048, nt * 16 + colT, 64, ks * 32 + lhalf);
        o = wmma_f16(a, b, o);
      }
      _Float16* op = OH + (mt * 16 + lhalf) * WA_DIM + h * 32 + nt * 16 + colT;
#pragma unroll
      for (int i = 0; i < 8; ++i) op[i * WA_DIM] = (_Float16)o[i];
    }
  }
  __syncthreads();

  // ---------- stage 5: out = OH @ proj_w^T + proj_b (fp32 store) ----------
#pragma unroll 1
  for (int j = 0; j < 4; ++j) {
    const int t  = wvs + 8 * j;        // scalar
    const int mt = t & 3, nt = t >> 2; // scalar
    v8f acc = {};
#pragma unroll
    for (int ks = 0; ks < 4; ++ks) {
      v16h a = f16_oper(OH, mt * 16 + colT, WA_DIM, ks * 32 + lhalf);
      v16h b;
      if constexpr (WF16) b = f16_oper(proj_w16, nt * 16 + colT, WA_DIM, ks * 32 + lhalf);
      else                b = glb_oper_f32(proj_w32, nt * 16 + colT, WA_DIM, ks * 32 + lhalf);
      acc = wmma_f16(a, b, acc);
    }
    const int col = nt * 16 + colT;
    const int m0  = mt * 16 + lhalf;
    float pb = proj_b[col];
    float* op = out + (size_t)bwin * (WA_N * WA_DIM) + m0 * WA_DIM + col;
    if (mt < 3) {      // scalar branch: all 8 rows valid
#pragma unroll
      for (int i = 0; i < 8; ++i) op[i * WA_DIM] = acc[i] + pb;
    } else {           // only rows < 49 stored
#pragma unroll
      for (int i = 0; i < 8; ++i)
        if (m0 + i < WA_N) op[i * WA_DIM] = acc[i] + pb;
    }
  }
}

extern "C" void kernel_launch(void* const* d_in, const int* in_sizes, int n_in,
                              void* d_out, int out_size, void* d_ws, size_t ws_size,
                              hipStream_t stream) {
  const float* x      = (const float*)d_in[0];
  const float* mask   = (const float*)d_in[1];
  const float* qkv_w  = (const float*)d_in[2];
  const float* qkv_b  = (const float*)d_in[3];
  const float* proj_w = (const float*)d_in[4];
  const float* proj_b = (const float*)d_in[5];
  const float* rtab   = (const float*)d_in[6];
  const int*   ridx   = (const int*)d_in[7];
  float* out = (float*)d_out;

  const int n_windows = in_sizes[0] / (WA_N * WA_DIM);   // B_ = 16384
  dim3 grid(n_windows), block(256);

  if (ws_size >= (size_t)WS_F16_BYTES) {
    _Float16* w16 = (_Float16*)d_ws;
    convert_weights_kernel<<<(WQKV_ELEMS + WPROJ_ELEMS) / (256 * 4), 256, 0, stream>>>(
        qkv_w, proj_w, w16);
    win_attn_fused_kernel<true><<<grid, block, WA_LDS_BYTES, stream>>>(
        x, mask, qkv_w, w16, qkv_b, proj_w, w16 + WQKV_ELEMS, proj_b, rtab, ridx, out);
  } else {
    win_attn_fused_kernel<false><<<grid, block, WA_LDS_BYTES, stream>>>(
        x, mask, qkv_w, nullptr, qkv_b, proj_w, nullptr, proj_b, rtab, ridx, out);
  }
}
